// HeadlessEdgeRegressionModel_3805341024627
// MI455X (gfx1250) — matrix operand order repaired and checked by reading, not compiled, not run on hardware
//
#include <hip/hip_runtime.h>
#include <hip/hip_bf16.h>

// ---------------------------------------------------------------------------
// Model dims (from reference)
// ---------------------------------------------------------------------------
#define NN 50000       // N_NODES
#define NE 800000      // N_EDGES
#define DN 256         // D_NODE
#define DE 64          // D_EDGE
#define FEM 128        // EMB == HID
#define NEG_SLOPE 0.2f
#define EPSV 1e-16f

typedef __attribute__((ext_vector_type(16))) __bf16 v16bf;
typedef __attribute__((ext_vector_type(8)))  __bf16 v8bf;
typedef __attribute__((ext_vector_type(8)))  float  v8f;

// ---------------------------------------------------------------------------
// Elementwise helpers
// ---------------------------------------------------------------------------
__global__ void cvt_f32_to_bf16(const float* __restrict__ in, __bf16* __restrict__ out, int n) {
    int i = blockIdx.x * blockDim.x + threadIdx.x;
    if (i < n) out[i] = (__bf16)in[i];
}

__global__ void fill_f32(float* p, float v, int n) {
    int i = blockIdx.x * blockDim.x + threadIdx.x;
    if (i < n) p[i] = v;
}

__global__ void fill_u32(unsigned* p, unsigned v, int n) {
    int i = blockIdx.x * blockDim.x + threadIdx.x;
    if (i < n) p[i] = v;
}

// ---------------------------------------------------------------------------
// Pack a f32 weight matrix B[K x 128] into per-lane WMMA B-fragments (bf16):
//   Bp[kt][nt][lane][0..15], kt = K/32 tiles, nt = 8 column tiles.
// Fragment layout (ISA 7.12.2): lane l (col = l&15, kq = 8*(l>=16)),
// element i -> B[kt*32 + kq + (i < 8 ? i : i + 8)][nt*16 + (l&15)].
// One block per kt (256 threads = 8 nt * 32 lanes).
// ---------------------------------------------------------------------------
__global__ void pack_b128(const float* __restrict__ B, __bf16* __restrict__ Bp) {
    int kt   = blockIdx.x;
    int nt   = (threadIdx.x >> 5) & 7;
    int lane = threadIdx.x & 31;
    int hi   = lane >> 4;
    int l15  = lane & 15;
    __bf16* d = Bp + (((size_t)kt * 8 + nt) * 32 + lane) * 16;
    const float* s = B + (size_t)(kt * 32 + hi * 8) * FEM + nt * 16 + l15;
#pragma unroll
    for (int i = 0; i < 16; ++i) d[i] = (__bf16)s[(size_t)(i < 8 ? i : i + 8) * FEM];
}

// ---------------------------------------------------------------------------
// bf16 WMMA GEMM, N = 128 fixed: out = act(A[MxK] @ B[Kx128] + bias).
// One wave computes a 16x128 output strip: A fragment loaded once per K-step
// (2x b128), fed to 8 WMMAs against 8 pre-packed B fragments (each a single
// contiguous 32-byte load). M % 16 == 0, K % 32 == 0.
// ---------------------------------------------------------------------------
__global__ __launch_bounds__(256)
void wmma_gemm(const __bf16* __restrict__ A, const __bf16* __restrict__ Bp,
               const float* __restrict__ bias,
               float* __restrict__ outF, __bf16* __restrict__ outB,
               int M, int K, int relu)
{
    int wave = (blockIdx.x * blockDim.x + threadIdx.x) >> 5;
    int lane = threadIdx.x & 31;
    if (wave * 16 >= M) return;                    // wave-uniform guard
    int hi  = lane >> 4;
    int l15 = lane & 15;

    const __bf16* arow = A + (size_t)(wave * 16 + l15) * K + hi * 8;
    v8f acc[8];
#pragma unroll
    for (int i = 0; i < 8; ++i) acc[i] = (v8f)0.0f;

    int KT = K >> 5;
    for (int kt = 0; kt < KT; ++kt) {
        v8bf a0 = *(const v8bf*)(arow + kt * 32);
        v8bf a1 = *(const v8bf*)(arow + kt * 32 + 16);
        v16bf a = __builtin_shufflevector(a0, a1, 0, 1, 2, 3, 4, 5, 6, 7,
                                          8, 9, 10, 11, 12, 13, 14, 15);
        __builtin_prefetch(arow + (kt + 2) * 32, 0, 0);   // global_prefetch_b8
        const __bf16* bbase = Bp + ((size_t)kt * 8 * 32 + lane) * 16;
#pragma unroll
        for (int nt = 0; nt < 8; ++nt) {
            v16bf b = *(const v16bf*)(bbase + (size_t)nt * 32 * 16);
            acc[nt] = __builtin_amdgcn_wmma_f32_16x16x32_bf16(false, a, false, b,
                                                              (short)0, acc[nt], false, false);
        }
    }
#pragma unroll
    for (int nt = 0; nt < 8; ++nt) {
        int col = nt * 16 + l15;
        float bv = bias ? bias[col] : 0.0f;
#pragma unroll
        for (int r = 0; r < 8; ++r) {
            float v = acc[nt][r] + bv;
            if (relu) v = fmaxf(v, 0.0f);
            size_t o = (size_t)(wave * 16 + r + hi * 8) * FEM + col;
            if (outF) outF[o] = v;
            if (outB) outB[o] = (__bf16)v;
        }
    }
}

// ---------------------------------------------------------------------------
// Edge-update GEMM with fused gather (K = 384 = 3 x 128 segments):
//   out[e] = concat(h[src[e]], h[dst[e]], eft[e]) @ B + bias
// Wave computes a 16x128 strip; per-row src/dst indices loaded once.
// ---------------------------------------------------------------------------
__global__ __launch_bounds__(256)
void edge_update_gemm(const __bf16* __restrict__ hft, const __bf16* __restrict__ eft,
                      const long long* __restrict__ src, const long long* __restrict__ dst,
                      const __bf16* __restrict__ Bp, const float* __restrict__ bias,
                      float* __restrict__ outF, __bf16* __restrict__ outB, int E)
{
    int wave = (blockIdx.x * blockDim.x + threadIdx.x) >> 5;
    int lane = threadIdx.x & 31;
    if (wave * 16 >= E) return;
    int hi  = lane >> 4;
    int l15 = lane & 15;

    int m = wave * 16 + l15;
    long long s = src[m], d = dst[m];
    const __bf16* seg[3];
    seg[0] = hft + (size_t)s * FEM + hi * 8;
    seg[1] = hft + (size_t)d * FEM + hi * 8;
    seg[2] = eft + (size_t)m * FEM + hi * 8;

    v8f acc[8];
#pragma unroll
    for (int i = 0; i < 8; ++i) acc[i] = (v8f)0.0f;

    for (int kt = 0; kt < 12; ++kt) {            // K = 384
        const __bf16* ap = seg[kt >> 2] + (kt & 3) * 32;
        v8bf a0 = *(const v8bf*)(ap);
        v8bf a1 = *(const v8bf*)(ap + 16);
        v16bf a = __builtin_shufflevector(a0, a1, 0, 1, 2, 3, 4, 5, 6, 7,
                                          8, 9, 10, 11, 12, 13, 14, 15);
        const __bf16* bbase = Bp + ((size_t)kt * 8 * 32 + lane) * 16;
#pragma unroll
        for (int nt = 0; nt < 8; ++nt) {
            v16bf b = *(const v16bf*)(bbase + (size_t)nt * 32 * 16);
            acc[nt] = __builtin_amdgcn_wmma_f32_16x16x32_bf16(false, a, false, b,
                                                              (short)0, acc[nt], false, false);
        }
    }
#pragma unroll
    for (int nt = 0; nt < 8; ++nt) {
        int col = nt * 16 + l15;
        float bv = bias[col];
#pragma unroll
        for (int r = 0; r < 8; ++r) {
            float v = acc[nt][r] + bv;
            size_t o = (size_t)(wave * 16 + r + hi * 8) * FEM + col;
            if (outF) outF[o] = v;
            if (outB) outB[o] = (__bf16)v;
        }
    }
}

// ---------------------------------------------------------------------------
// Row-wise matvec over 128-wide rows: out[r] = dot(X[r,:], v). Wave per row.
// ---------------------------------------------------------------------------
template <typename TIn>
__global__ void matvec128(const TIn* __restrict__ X, const float* __restrict__ v,
                          float* __restrict__ out, int rows)
{
    int wave = (blockIdx.x * blockDim.x + threadIdx.x) >> 5;
    int lane = threadIdx.x & 31;
    if (wave >= rows) return;
    const TIn* xp = X + (size_t)wave * FEM;
    float s = 0.0f;
#pragma unroll
    for (int i = 0; i < 4; ++i) {
        int c = lane + 32 * i;
        s += (float)xp[c] * v[c];
    }
#pragma unroll
    for (int off = 16; off > 0; off >>= 1) s += __shfl_down(s, off, 32);
    if (lane == 0) out[wave] = s;
}

// ---------------------------------------------------------------------------
// GAT segment kernels
// ---------------------------------------------------------------------------
__global__ void deg_kernel(const long long* __restrict__ dst, float* __restrict__ deg, int E) {
    int i = blockIdx.x * blockDim.x + threadIdx.x;
    if (i < E) atomicAdd(deg + dst[i], 1.0f);
}

__global__ void segsum_edge(const __bf16* __restrict__ eft, const long long* __restrict__ dst,
                            float* __restrict__ acc, int E)
{
    int wave = (blockIdx.x * blockDim.x + threadIdx.x) >> 5;
    int lane = threadIdx.x & 31;
    if (wave >= E) return;
    long long d = dst[wave];
    const __bf16* ep = eft + (size_t)wave * FEM;
    float* op = acc + (size_t)d * FEM;
#pragma unroll
    for (int i = 0; i < 4; ++i) {
        int c = lane + 32 * i;
        atomicAdd(op + c, (float)ep[c]);
    }
}

__global__ void div_deg(float* __restrict__ acc, const float* __restrict__ deg, int n) {
    int i = blockIdx.x * blockDim.x + threadIdx.x;
    if (i < n) acc[i] /= fmaxf(deg[i >> 7], 1.0f);
}

__global__ void alpha_kernel(const float* __restrict__ asrc, const float* __restrict__ adst,
                             float* __restrict__ aedge,  // in: aedge, out: alpha (in place)
                             const long long* __restrict__ src, const long long* __restrict__ dst,
                             int E, int Ntot, unsigned* __restrict__ amax)
{
    int i = blockIdx.x * blockDim.x + threadIdx.x;
    if (i >= Ntot) return;
    long long s, d;
    if (i < E) { s = src[i]; d = dst[i]; } else { s = d = i - E; }
    float a = asrc[s] + adst[d] + aedge[i];
    a = a > 0.0f ? a : a * NEG_SLOPE;
    aedge[i] = a;
    unsigned bits = __float_as_uint(a);
    unsigned ord  = bits ^ ((unsigned)((int)bits >> 31) | 0x80000000u);
    atomicMax(amax + d, ord);
}

__global__ void ea_kernel(float* __restrict__ alpha,  // in: alpha, out: ea (in place)
                          const unsigned* __restrict__ amax,
                          const long long* __restrict__ dst,
                          int E, int Ntot, float* __restrict__ denom)
{
    int i = blockIdx.x * blockDim.x + threadIdx.x;
    if (i >= Ntot) return;
    long long d = (i < E) ? dst[i] : (long long)(i - E);
    unsigned ord  = amax[d];
    unsigned bits = (ord & 0x80000000u) ? (ord ^ 0x80000000u) : ~ord;
    float mx = __uint_as_float(bits);
    float e  = __expf(alpha[i] - mx);
    alpha[i] = e;
    atomicAdd(denom + d, e);
}

__global__ void scatter_kernel(const float* __restrict__ ea, const float* __restrict__ denom,
                               const __bf16* __restrict__ hW,
                               const long long* __restrict__ src, const long long* __restrict__ dst,
                               int E, int Ntot, float* __restrict__ outacc)
{
    int wave = (blockIdx.x * blockDim.x + threadIdx.x) >> 5;
    int lane = threadIdx.x & 31;
    if (wave >= Ntot) return;
    long long s, d;
    if (wave < E) { s = src[wave]; d = dst[wave]; } else { s = d = wave - E; }
    float coef = ea[wave] / (denom[d] + EPSV);
    const __bf16* hp = hW + (size_t)s * FEM;
    float* op = outacc + (size_t)d * FEM;
#pragma unroll
    for (int i = 0; i < 4; ++i) {
        int c = lane + 32 * i;
        atomicAdd(op + c, coef * (float)hp[c]);
    }
}

__global__ void bias_finish(const float* __restrict__ acc, const float* __restrict__ bias,
                            __bf16* __restrict__ outB, float* __restrict__ outF, int n)
{
    int i = blockIdx.x * blockDim.x + threadIdx.x;
    if (i >= n) return;
    float v = acc[i] + bias[i & (FEM - 1)];
    outB[i] = (__bf16)v;
    if (outF) outF[i] = v;
}

// ---------------------------------------------------------------------------
// Host orchestration
// ---------------------------------------------------------------------------
static inline size_t al256(size_t x) { return (x + 255) & ~(size_t)255; }

extern "C" void kernel_launch(void* const* d_in, const int* in_sizes, int n_in,
                              void* d_out, int out_size, void* d_ws, size_t ws_size,
                              hipStream_t stream)
{
    (void)in_sizes; (void)n_in; (void)out_size; (void)ws_size;
    const float*     x_f   = (const float*)d_in[0];
    const long long* ei    = (const long long*)d_in[1];
    const long long* src   = ei;
    const long long* dst   = ei + NE;
    const float*     ea_f  = (const float*)d_in[2];
    const float* nW1 = (const float*)d_in[3];  const float* nb1 = (const float*)d_in[4];
    const float* nW2 = (const float*)d_in[5];  const float* nb2 = (const float*)d_in[6];
    const float* eW1 = (const float*)d_in[7];  const float* eb1 = (const float*)d_in[8];
    const float* eW2 = (const float*)d_in[9];  const float* eb2 = (const float*)d_in[10];
    const float* gW[2]  = {(const float*)d_in[11], (const float*)d_in[19]};
    const float* gb[2]  = {(const float*)d_in[12], (const float*)d_in[20]};
    const float* gas[2] = {(const float*)d_in[13], (const float*)d_in[21]};
    const float* gad[2] = {(const float*)d_in[14], (const float*)d_in[22]};
    const float* gWe[2] = {(const float*)d_in[15], (const float*)d_in[23]};
    const float* gae[2] = {(const float*)d_in[16], (const float*)d_in[24]};
    const float* uW[2]  = {(const float*)d_in[17], (const float*)d_in[25]};
    const float* ub[2]  = {(const float*)d_in[18], (const float*)d_in[26]};

    float* out_h = (float*)d_out;                      // [NN x 128]
    float* out_e = (float*)d_out + (size_t)NN * FEM;   // [NE x 128]

    // ---- workspace carve-up ----
    char* ws = (char*)d_ws; size_t off = 0;
    auto carve = [&](size_t bytes) { char* p = ws + off; off += al256(bytes); return p; };
    __bf16* xb     = (__bf16*)carve((size_t)NN * DN * 2);
    __bf16* eab    = (__bf16*)carve((size_t)NE * DE * 2);
    __bf16* h_a    = (__bf16*)carve((size_t)NN * FEM * 2);
    __bf16* h_b    = (__bf16*)carve((size_t)NN * FEM * 2);
    __bf16* e_a    = (__bf16*)carve((size_t)NE * FEM * 2);
    __bf16* e_b    = (__bf16*)carve((size_t)NE * FEM * 2);
    __bf16* hWb    = (__bf16*)carve((size_t)NN * FEM * 2);
    float*  loopat = (float*)carve((size_t)NN * FEM * 4);
    float*  outacc = (float*)carve((size_t)NN * FEM * 4);
    float*  deg    = (float*)carve((size_t)NN * 4);
    float*  asrc   = (float*)carve((size_t)NN * 4);
    float*  adst   = (float*)carve((size_t)NN * 4);
    float*  alphab = (float*)carve((size_t)(NE + NN) * 4);
    unsigned* amax = (unsigned*)carve((size_t)NN * 4);
    float*  denom  = (float*)carve((size_t)NN * 4);
    float*  we_ae  = (float*)carve((size_t)FEM * 4);
    // pre-packed bf16 weight fragments (element count == K*128 each)
    __bf16* nW1p = (__bf16*)carve((size_t)DN * FEM * 2);
    __bf16* nW2p = (__bf16*)carve((size_t)FEM * FEM * 2);
    __bf16* eW1p = (__bf16*)carve((size_t)DE * FEM * 2);
    __bf16* eW2p = (__bf16*)carve((size_t)FEM * FEM * 2);
    __bf16* gWp[2], *uWp[2];
    for (int i = 0; i < 2; ++i) {
        gWp[i] = (__bf16*)carve((size_t)FEM * FEM * 2);
        uWp[i] = (__bf16*)carve((size_t)3 * FEM * FEM * 2);
    }

    const int TB = 256;
    auto gthr = [&](int n) { return dim3((n + TB - 1) / TB); };   // thread-per-elem
    auto gwav = [&](int n) { return dim3((n + 7) / 8); };         // wave-per-row
    auto ggem = [&](int M) { return dim3((M / 16 + 7) / 8); };    // wave-per-16-row strip

    // ---- precision conversion + weight packing ----
    cvt_f32_to_bf16<<<gthr(NN * DN), TB, 0, stream>>>(x_f, xb, NN * DN);
    cvt_f32_to_bf16<<<gthr(NE * DE), TB, 0, stream>>>(ea_f, eab, NE * DE);
    pack_b128<<<DN / 32, TB, 0, stream>>>(nW1, nW1p);
    pack_b128<<<FEM / 32, TB, 0, stream>>>(nW2, nW2p);
    pack_b128<<<DE / 32, TB, 0, stream>>>(eW1, eW1p);
    pack_b128<<<FEM / 32, TB, 0, stream>>>(eW2, eW2p);
    for (int i = 0; i < 2; ++i) {
        pack_b128<<<FEM / 32, TB, 0, stream>>>(gW[i], gWp[i]);
        pack_b128<<<3 * FEM / 32, TB, 0, stream>>>(uW[i], uWp[i]);
    }

    // ---- node MLP ----
    wmma_gemm<<<ggem(NN), TB, 0, stream>>>(xb, nW1p, nb1, nullptr, h_b, NN, DN, 1);
    wmma_gemm<<<ggem(NN), TB, 0, stream>>>(h_b, nW2p, nb2, nullptr, h_a, NN, FEM, 1);
    // ---- edge MLP ----
    wmma_gemm<<<ggem(NE), TB, 0, stream>>>(eab, eW1p, eb1, nullptr, e_b, NE, DE, 1);
    wmma_gemm<<<ggem(NE), TB, 0, stream>>>(e_b, eW2p, eb2, nullptr, e_a, NE, FEM, 1);

    __bf16* h_cur = h_a; __bf16* h_nxt = h_b;
    __bf16* e_cur = e_a; __bf16* e_nxt = e_b;
    const int NTOT = NE + NN;

    for (int L = 0; L < 2; ++L) {
        bool last = (L == 1);
        fill_f32<<<gthr(NN), TB, 0, stream>>>(deg, 0.0f, NN);
        fill_f32<<<gthr(NN * FEM), TB, 0, stream>>>(loopat, 0.0f, NN * FEM);
        fill_f32<<<gthr(NN * FEM), TB, 0, stream>>>(outacc, 0.0f, NN * FEM);
        fill_u32<<<gthr(NN), TB, 0, stream>>>(amax, 0u, NN);
        fill_f32<<<gthr(NN), TB, 0, stream>>>(denom, 0.0f, NN);

        // self-loop attr = mean of incoming edge features
        deg_kernel<<<gthr(NE), TB, 0, stream>>>(dst, deg, NE);
        segsum_edge<<<gwav(NE), TB, 0, stream>>>(e_cur, dst, loopat, NE);
        div_deg<<<gthr(NN * FEM), TB, 0, stream>>>(loopat, deg, NN * FEM);

        // h @ W (no bias, no relu) -> bf16
        wmma_gemm<<<ggem(NN), TB, 0, stream>>>(h_cur, gWp[L], nullptr, nullptr, hWb,
                                               NN, FEM, 0);
        // attention logits: alpha_src/dst per node, alpha_edge = e2 @ (We @ a_e)
        matvec128<__bf16><<<gwav(NN), TB, 0, stream>>>(hWb, gas[L], asrc, NN);
        matvec128<__bf16><<<gwav(NN), TB, 0, stream>>>(hWb, gad[L], adst, NN);
        matvec128<float><<<gwav(FEM), TB, 0, stream>>>(gWe[L], gae[L], we_ae, FEM);
        matvec128<__bf16><<<gwav(NE), TB, 0, stream>>>(e_cur, we_ae, alphab, NE);
        matvec128<float><<<gwav(NN), TB, 0, stream>>>(loopat, we_ae, alphab + NE, NN);

        // segment softmax + aggregate
        alpha_kernel<<<gthr(NTOT), TB, 0, stream>>>(asrc, adst, alphab, src, dst, NE, NTOT, amax);
        ea_kernel<<<gthr(NTOT), TB, 0, stream>>>(alphab, amax, dst, NE, NTOT, denom);
        scatter_kernel<<<gwav(NTOT), TB, 0, stream>>>(alphab, denom, hWb, src, dst, NE, NTOT, outacc);
        bias_finish<<<gthr(NN * FEM), TB, 0, stream>>>(outacc, gb[L], h_nxt,
                                                       last ? out_h : nullptr, NN * FEM);

        // edge update: concat(h[src], h[dst], e) @ uW + ub  (gather fused into GEMM)
        edge_update_gemm<<<ggem(NE), TB, 0, stream>>>(
            h_nxt, e_cur, src, dst, uWp[L], ub[L],
            last ? out_e : nullptr, last ? nullptr : e_nxt, NE);

        __bf16* t;
        t = h_cur; h_cur = h_nxt; h_nxt = t;
        t = e_cur; e_cur = e_nxt; e_nxt = t;
    }
}
